// TTMiniMaxM1Attention_1657857376612
// MI455X (gfx1250) — matrix operand order repaired and checked by reading, not compiled, not run on hardware
//
#include <hip/hip_runtime.h>
#include <hip/hip_bf16.h>

// ---------------------------------------------------------------------------
// MiniMax-M1 GQA attention layer for MI455X (gfx1250, wave32, WMMA).
// B=2 S=2048 HID=2048 NH=32 KVH=8 HD=64 GROUPS=4
// Pipeline: 3x GEMM (bf16 WMMA, f32 accum, double-buffered LDS, prefetch)
// -> RoPE pack -> transposed flash attention (both matmuls as WMMA with
// direct-loaded K/Q/V operands and one xor-16 shuffle for P^T, exp2-domain online softmax) -> output GEMM.
// ---------------------------------------------------------------------------

typedef __bf16 bf16_t;
typedef __attribute__((ext_vector_type(16))) __bf16 bf16x16;
typedef __attribute__((ext_vector_type(8)))  float  f32x8;

union FragU { bf16x16 v; uint4 q[2]; };

__device__ static inline bf16_t f2bf(float f) {
  unsigned u = __float_as_uint(f);
  u += 0x7FFFu + ((u >> 16) & 1u);           // round-to-nearest-even
  unsigned short s = (unsigned short)(u >> 16);
  return __builtin_bit_cast(bf16_t, s);
}
__device__ static inline unsigned short f2bfu(float f) {
  return __builtin_bit_cast(unsigned short, f2bf(f));
}

__device__ static inline f32x8 wmma_bf16(bf16x16 a, bf16x16 b, f32x8 c) {
  // emits v_wmma_f32_16x16x32_bf16
  return __builtin_amdgcn_wmma_f32_16x16x32_bf16(
      /*neg_a=*/false, a, /*neg_b=*/false, b,
      /*c_mod=*/(short)0, c, /*reuse_a=*/false, /*reuse_b=*/false);
}

#define S_LEN   2048
#define HID     2048
#define NH      32
#define KVH     8
#define HD      64
#define GROUPS  4
#define LOG2E   1.44269504088896340736f

// ---------------------------------------------------------------------------
// GEMM: C[M,N] (f32) = A[M,K] * W[K,N].  A is f32 (converted) or bf16.
// Block = 256 thr (8 waves), tile 128x128, K-step 32, double-buffered LDS.
// Requires M%128==0, N%128==0, K%32==0 (true for all uses here).
// ---------------------------------------------------------------------------
template <bool ABF16>
__global__ __launch_bounds__(256)
void gemm_bf16wmma(const void* __restrict__ Aptr, const float* __restrict__ W,
                   float* __restrict__ C, int M, int N, int K) {
  __shared__ bf16_t sA[2][128][40];   // [buf][m][k], row pitch 80B (16B align)
  __shared__ bf16_t sB[2][128][40];   // [buf][n][k] (W transposed)

  const int tid    = threadIdx.x;
  const int lane   = tid & 31;
  const int wave   = tid >> 5;
  const int lane16 = lane & 15;
  const int hv     = lane >> 4;          // half of wave
  const int wm     = wave & 1;           // 0..1 -> 64-row slab
  const int wn     = wave >> 1;          // 0..3 -> 32-col slab
  const int M0     = blockIdx.y * 128;
  const int N0     = blockIdx.x * 128;

  // A loader: row = tid/2 (0..127), 16 consecutive k
  const int ar = tid >> 1;
  const int ak = (tid & 1) * 16;
  // W loader: k = tid/8 (0..31), 16 consecutive n
  const int wk  = tid >> 3;
  const int wn0 = (tid & 7) * 16;

  const float*          Af = (const float*)Aptr;
  const unsigned short* Ab = (const unsigned short*)Aptr;

  f32x8 acc[4][2];
#pragma unroll
  for (int i = 0; i < 4; ++i)
#pragma unroll
    for (int j = 0; j < 2; ++j) acc[i][j] = (f32x8)0.0f;

  bf16_t regA[16], regB[16];

  auto stage_global = [&](int k0) {
    if constexpr (ABF16) {
      const unsigned short* src = Ab + (size_t)(M0 + ar) * K + k0 + ak;
#pragma unroll
      for (int i = 0; i < 16; ++i) regA[i] = __builtin_bit_cast(bf16_t, src[i]);
    } else {
      const float* src = Af + (size_t)(M0 + ar) * K + k0 + ak;
#pragma unroll
      for (int i = 0; i < 16; ++i) regA[i] = f2bf(src[i]);
    }
    const float* wsrc = W + (size_t)(k0 + wk) * N + N0 + wn0;
#pragma unroll
    for (int i = 0; i < 16; ++i) regB[i] = f2bf(wsrc[i]);
  };
  auto stage_lds = [&](int buf) {
#pragma unroll
    for (int i = 0; i < 16; ++i) sA[buf][ar][ak + i] = regA[i];
#pragma unroll
    for (int i = 0; i < 16; ++i) sB[buf][wn0 + i][wk] = regB[i];
  };

  stage_global(0);
  stage_lds(0);
  __syncthreads();

  int buf = 0;
  for (int k0 = 0; k0 < K; k0 += 32) {
    const bool last = (k0 + 32 >= K);
    if (!last) {
      stage_global(k0 + 32);               // overlap with WMMAs below
      if (k0 + 64 < K) {                   // pull tile i+2 toward L2/L0
        if constexpr (ABF16)
          __builtin_prefetch(Ab + (size_t)(M0 + ar) * K + k0 + 64 + ak, 0, 3);
        else
          __builtin_prefetch(Af + (size_t)(M0 + ar) * K + k0 + 64 + ak, 0, 3);
        __builtin_prefetch(W + (size_t)(k0 + 64 + wk) * N + N0 + wn0, 0, 3);
      }
    }

    FragU afr[4], bfr[2];
#pragma unroll
    for (int mt = 0; mt < 4; ++mt) {
      const int m = wm * 64 + mt * 16 + lane16;
      afr[mt].q[0] = *(const uint4*)&sA[buf][m][8 * hv];
      afr[mt].q[1] = *(const uint4*)&sA[buf][m][16 + 8 * hv];
    }
#pragma unroll
    for (int nt = 0; nt < 2; ++nt) {
      const int n = wn * 32 + nt * 16 + lane16;
      bfr[nt].q[0] = *(const uint4*)&sB[buf][n][8 * hv];
      bfr[nt].q[1] = *(const uint4*)&sB[buf][n][16 + 8 * hv];
    }
#pragma unroll
    for (int mt = 0; mt < 4; ++mt)
#pragma unroll
      for (int nt = 0; nt < 2; ++nt)
        acc[mt][nt] = wmma_bf16(afr[mt].v, bfr[nt].v, acc[mt][nt]);

    if (!last) {
      stage_lds(buf ^ 1);
      __syncthreads();
      buf ^= 1;
    }
  }

#pragma unroll
  for (int mt = 0; mt < 4; ++mt)
#pragma unroll
    for (int nt = 0; nt < 2; ++nt) {
      const int row0 = M0 + wm * 64 + mt * 16 + 8 * hv;
      const int col  = N0 + wn * 32 + nt * 16 + lane16;
      float* outp = C + (size_t)row0 * N + col;
#pragma unroll
      for (int r = 0; r < 8; ++r) outp[(size_t)r * N] = acc[mt][nt][r];
    }
}

// ---------------------------------------------------------------------------
// RoPE pack kernels (f32 -> bf16).  inv_freq = 2^(-d2 * log2(1e4)/32)
// ---------------------------------------------------------------------------
#define LOG2_1E4_OVER_32 0.41524101186092026f  // log2(10000)/32

__global__ __launch_bounds__(256)
void rope_q_kernel(const float* __restrict__ Qt, const int* __restrict__ pos,
                   unsigned short* __restrict__ Qb) {
  const int idx = blockIdx.x * 256 + threadIdx.x;  // B*S*NH*32
  const int d2  = idx & 31;
  const int t   = idx >> 5;
  const int hh  = t & 31;          // head
  const int row = t >> 5;          // b*S + s
  const float p    = (float)pos[row];
  const float invf = exp2f(-(float)d2 * LOG2_1E4_OVER_32);
  float sn, cs;
  __sincosf(p * invf, &sn, &cs);
  const size_t base = (size_t)row * 2048 + hh * 64 + d2;
  const float q1 = Qt[base], q2 = Qt[base + 32];
  Qb[base]      = f2bfu(q1 * cs - q2 * sn);
  Qb[base + 32] = f2bfu(q2 * cs + q1 * sn);
}

__global__ __launch_bounds__(256)
void rope_k_kernel(const float* __restrict__ Kt, const int* __restrict__ pos,
                   unsigned short* __restrict__ Kb) {
  const int idx = blockIdx.x * 256 + threadIdx.x;  // B*S*KVH*32
  const int d2  = idx & 31;
  const int t   = idx >> 5;
  const int hh  = t & 7;           // kv head
  const int row = t >> 3;          // b*S + s
  const int b   = row >> 11;       // S = 2048
  const int s   = row & 2047;
  const float p    = (float)pos[row];
  const float invf = exp2f(-(float)d2 * LOG2_1E4_OVER_32);
  float sn, cs;
  __sincosf(p * invf, &sn, &cs);
  const size_t in_base  = (size_t)row * 512 + hh * 64 + d2;
  const size_t out_base = ((size_t)(b * KVH + hh) * S_LEN + s) * HD + d2;
  const float k1 = Kt[in_base], k2 = Kt[in_base + 32];
  Kb[out_base]      = f2bfu(k1 * cs - k2 * sn);
  Kb[out_base + 32] = f2bfu(k2 * cs + k1 * sn);
}

// V: [B,S,KVH*HD] f32 -> [B,KVH,HD,S] bf16 (dim-major so V^T rows are
// contiguous over keys == this code's contiguous-k A-fragment layout)
__global__ __launch_bounds__(256)
void pack_v_kernel(const float* __restrict__ Vt_in,
                   unsigned short* __restrict__ Vt_out) {
  const int idx  = blockIdx.x * 256 + threadIdx.x;  // B*KVH*HD*S
  const int s    = idx & 2047;
  const int t    = idx >> 11;
  const int d    = t & 63;
  const int t2   = t >> 6;
  const int hh   = t2 & 7;
  const int b    = t2 >> 3;
  const float v = Vt_in[((size_t)(b * S_LEN + s)) * 512 + hh * 64 + d];
  Vt_out[((size_t)(b * KVH + hh) * HD + d) * S_LEN + s] = f2bfu(v);
}

// ---------------------------------------------------------------------------
// Transposed flash attention. One wave = 16 queries, block = 8 waves = 128
// queries. S^T = K*Q^T (A from K rows, B from Q rows), out^T = V^T * P^T.
// Softmax stats are per-lane scalars (lane16 == query column); softmax is
// computed in the exp2 domain (base change is a no-op for softmax).
// ---------------------------------------------------------------------------
__global__ __launch_bounds__(256)
void attn_kernel(const unsigned short* __restrict__ Qb,
                 const unsigned short* __restrict__ Kb,
                 const unsigned short* __restrict__ Vt,
                 unsigned short* __restrict__ O) {
  const int tid    = threadIdx.x;
  const int lane   = tid & 31;
  const int wave   = tid >> 5;
  const int lane16 = lane & 15;
  const int hv     = lane >> 4;
  const int h   = blockIdx.y;          // head
  const int b   = blockIdx.z;
  const int hkv = h >> 2;              // GROUPS = 4
  const int q0  = blockIdx.x * 128 + wave * 16;
  const int q   = q0 + lane16;         // this lane's query column

  const unsigned short* qrow =
      Qb + (size_t)(b * S_LEN + q) * 2048 + h * 64;
  const unsigned short* Kbase =
      Kb + (size_t)(b * KVH + hkv) * S_LEN * HD;
  const unsigned short* Vbase =
      Vt + (size_t)(b * KVH + hkv) * HD * S_LEN;

  // Q^T B-fragments (hd 0..31 and 32..63): contiguous per lane
  FragU bq0, bq1;
  bq0.q[0] = *(const uint4*)(qrow + 16 * hv);
  bq0.q[1] = *(const uint4*)(qrow + 16 * hv + 8);
  bq1.q[0] = *(const uint4*)(qrow + 32 + 16 * hv);
  bq1.q[1] = *(const uint4*)(qrow + 32 + 16 * hv + 8);

  f32x8 acc[4];
#pragma unroll
  for (int i = 0; i < 4; ++i) acc[i] = (f32x8)0.0f;
  float m_run = -1e30f, l_run = 0.0f;   // exp2-domain stats

  const float SCALE2 = 0.125f * LOG2E;  // (1/sqrt(HD)) * log2(e)

  for (int kbase = 0; kbase < q0 + 16; kbase += 32) {
    if (kbase + 32 < q0 + 16) {          // prefetch next key block
      __builtin_prefetch(Kbase + (size_t)(kbase + 32 + lane16) * HD, 0, 3);
      __builtin_prefetch(Kbase + (size_t)(kbase + 48 + lane16) * HD, 0, 3);
    }

    // --- scores: two 16-key tiles of S^T ---
    float sc[2][8];
#pragma unroll
    for (int t = 0; t < 2; ++t) {
      const int key_m = kbase + t * 16 + lane16;       // A row = key
      const unsigned short* krow = Kbase + (size_t)key_m * HD;
      FragU ka0, ka1;
      ka0.q[0] = *(const uint4*)(krow + 16 * hv);
      ka0.q[1] = *(const uint4*)(krow + 16 * hv + 8);
      ka1.q[0] = *(const uint4*)(krow + 32 + 16 * hv);
      ka1.q[1] = *(const uint4*)(krow + 32 + 16 * hv + 8);
      f32x8 s = (f32x8)0.0f;
      s = wmma_bf16(ka0.v, bq0.v, s);
      s = wmma_bf16(ka1.v, bq1.v, s);
#pragma unroll
      for (int r = 0; r < 8; ++r) {
        const int key = kbase + t * 16 + r + 8 * hv;
        float x = s[r] * SCALE2;                       // exp2 domain
        if (key > q) x += -1e9f;                       // causal mask
        sc[t][r] = x;
      }
    }

    // --- online softmax (stats per lane == per query) ---
    float m_loc = -1e30f;
#pragma unroll
    for (int t = 0; t < 2; ++t)
#pragma unroll
      for (int r = 0; r < 8; ++r) m_loc = fmaxf(m_loc, sc[t][r]);
    m_loc = fmaxf(m_loc, __shfl_xor(m_loc, 16, 32));
    const float m_new = fmaxf(m_run, m_loc);

    float p[2][8];
    float l_tile = 0.0f;
#pragma unroll
    for (int t = 0; t < 2; ++t)
#pragma unroll
      for (int r = 0; r < 8; ++r) {
        p[t][r] = exp2f(sc[t][r] - m_new);             // v_exp_f32 direct
        l_tile += p[t][r];
      }
    l_tile += __shfl_xor(l_tile, 16, 32);

    const float alpha = exp2f(m_run - m_new);
    l_run = l_run * alpha + l_tile;
    m_run = m_new;
#pragma unroll
    for (int dt = 0; dt < 4; ++dt) acc[dt] = acc[dt] * alpha;

    // --- build P^T B-fragment (32 keys x 16 queries): one xor-16 shuffle ---
    FragU bp;
#pragma unroll
    for (int r = 0; r < 8; ++r) {
      const float send = (hv == 0) ? p[1][r] : p[0][r];
      const float recv = __shfl_xor(send, 16, 32);
      const float lo = (hv == 0) ? p[0][r] : recv;   // keys 16*hv + 0..7
      const float hi = (hv == 0) ? recv : p[1][r];   // keys 16*hv + 8..15
      bp.v[r]     = f2bf(lo);
      bp.v[8 + r] = f2bf(hi);
    }

    // --- out^T += V^T * P^T (4 dim tiles) ---
#pragma unroll
    for (int dt = 0; dt < 4; ++dt) {
      const int d = dt * 16 + lane16;                 // A row = dim
      const unsigned short* vrow = Vbase + (size_t)d * S_LEN + kbase;
      FragU va;
      va.q[0] = *(const uint4*)(vrow + 16 * hv);
      va.q[1] = *(const uint4*)(vrow + 16 * hv + 8);
      acc[dt] = wmma_bf16(va.v, bp.v, acc[dt]);
    }
  }

  const float inv_l = 1.0f / l_run;
  unsigned short* orow = O + (size_t)(b * S_LEN + q) * 2048 + h * 64;
#pragma unroll
  for (int dt = 0; dt < 4; ++dt)
#pragma unroll
    for (int r = 0; r < 8; ++r) {
      const int d = dt * 16 + r + 8 * hv;
      orow[d] = f2bfu(acc[dt][r] * inv_l);
    }
}

// ---------------------------------------------------------------------------
// Host launcher
// ---------------------------------------------------------------------------
extern "C" void kernel_launch(void* const* d_in, const int* in_sizes, int n_in,
                              void* d_out, int out_size, void* d_ws,
                              size_t ws_size, hipStream_t stream) {
  const float* hs  = (const float*)d_in[0];
  // d_in[1] = attention_mask: causal -1e9 mask, reproduced analytically
  const int*   pos = (const int*)d_in[2];
  const float* Wq  = (const float*)d_in[3];
  const float* Wk  = (const float*)d_in[4];
  const float* Wv  = (const float*)d_in[5];
  const float* Wo  = (const float*)d_in[6];
  float*       out = (float*)d_out;

  char* ws = (char*)d_ws;
  float* Qtmp = (float*)ws;                 ws += (size_t)4096 * 2048 * 4;  // 33.5MB
  float* Ktmp = (float*)ws;                 ws += (size_t)4096 * 512 * 4;   //  8.4MB
  float* Vtmp = (float*)ws;                 ws += (size_t)4096 * 512 * 4;   //  8.4MB
  unsigned short* Qb   = (unsigned short*)ws; ws += (size_t)4096 * 2048 * 2; // 16.8MB
  unsigned short* Kb   = (unsigned short*)ws; ws += (size_t)2 * 8 * 2048 * 64 * 2; // 4.2MB
  unsigned short* Vtb  = (unsigned short*)ws; ws += (size_t)2 * 8 * 64 * 2048 * 2; // 4.2MB
  unsigned short* Attn = (unsigned short*)ws; ws += (size_t)4096 * 2048 * 2; // 16.8MB

  dim3 blk(256);

  // QKV projections (bf16 WMMA, f32 accumulate)
  gemm_bf16wmma<false><<<dim3(16, 32), blk, 0, stream>>>(hs, Wq, Qtmp, 4096, 2048, 2048);
  gemm_bf16wmma<false><<<dim3(4, 32),  blk, 0, stream>>>(hs, Wk, Ktmp, 4096, 512, 2048);
  gemm_bf16wmma<false><<<dim3(4, 32),  blk, 0, stream>>>(hs, Wv, Vtmp, 4096, 512, 2048);

  // RoPE + bf16 pack
  rope_q_kernel<<<16384, blk, 0, stream>>>(Qtmp, pos, Qb);
  rope_k_kernel<<<4096,  blk, 0, stream>>>(Ktmp, pos, Kb);
  pack_v_kernel<<<8192,  blk, 0, stream>>>(Vtmp, Vtb);

  // Flash attention: grid = (q-tiles, heads, batch)
  attn_kernel<<<dim3(16, 32, 2), blk, 0, stream>>>(Qb, Kb, Vtb, Attn);

  // Output projection
  gemm_bf16wmma<true><<<dim3(16, 32), blk, 0, stream>>>(Attn, Wo, out, 4096, 2048, 2048);
}
